// SubmanifoldSparseConv_16441134809108
// MI455X (gfx1250) — matrix operand order repaired and checked by reading, not compiled, not run on hardware
//
#include <hip/hip_runtime.h>

typedef __attribute__((ext_vector_type(16))) _Float16 v16h;
typedef __attribute__((ext_vector_type(8)))  _Float16 v8h;
typedef __attribute__((ext_vector_type(8)))  float    v8f;

#define KEYB   130           // key base (G + 2)
#define NOFF   27
#define CIN    32
#define COUT   32
#define WPACK_ELEMS (NOFF * 2 * 32 * 16)   // 27648 f16 fragment elements

// ---------------------------------------------------------------------------
// Hash table: open addressing, linear probe. Value = MIN original index of any
// point in that voxel (matches jnp.argsort stability in the reference).
// ---------------------------------------------------------------------------
__device__ __forceinline__ int hash_lookup(const int* __restrict__ hkey,
                                           const int* __restrict__ hval,
                                           int kq, unsigned hmask, int hshift) {
    unsigned s = ((unsigned)kq * 2654435761u) >> hshift;
    for (;;) {
        int kk = hkey[s];
        if (kk == kq) return hval[s];
        if (kk == -1) return -1;
        s = (s + 1) & hmask;
    }
}

__global__ void init_hash_kernel(int* hkey, int* hval, int hcap) {
    int i = blockIdx.x * blockDim.x + threadIdx.x;
    if (i < hcap) { hkey[i] = -1; hval[i] = 0x7fffffff; }
}

__global__ void build_hash_kernel(const int* __restrict__ posns,
                                  int* hkey, int* hval,
                                  int n, unsigned hmask, int hshift) {
    int i = blockIdx.x * blockDim.x + threadIdx.x;
    if (i >= n) return;
    int x = posns[i * 3 + 0], y = posns[i * 3 + 1], z = posns[i * 3 + 2];
    int k = ((x + 1) * KEYB + (y + 1)) * KEYB + (z + 1);
    unsigned s = ((unsigned)k * 2654435761u) >> hshift;
    for (;;) {
        int prev = atomicCAS(&hkey[s], -1, k);
        if (prev == -1 || prev == k) { atomicMin(&hval[s], i); break; }
        s = (s + 1) & hmask;
    }
}

// f32 -> f16 feature table with an appended ZERO row at index n
// (missing neighbors gather from row n -> contribute exactly 0; branchless loop)
__global__ void cvt_feat_kernel(const float* __restrict__ src,
                                _Float16* __restrict__ dst,
                                int ntotal, int nsrc) {
    int i = blockIdx.x * blockDim.x + threadIdx.x;
    if (i < ntotal) dst[i] = (i < nsrc) ? (_Float16)src[i] : (_Float16)0.0f;
}

// Pack weights [27][Cin=32][Cout=32] f32 into exact B-fragment layout:
//   wpack[((o*2+ct)*32 + lane)*16 + e] = f16( W[o][K][N] )
//   K = (e&7) + 8*(lane>=16) + 16*(e>=8),  N = (lane&15) + 16*ct
__global__ void pack_w_kernel(const float* __restrict__ w,
                              _Float16* __restrict__ wpack) {
    int idx = blockIdx.x * blockDim.x + threadIdx.x;
    if (idx >= WPACK_ELEMS) return;
    int e    = idx & 15;
    int lane = (idx >> 4) & 31;
    int ct   = (idx >> 9) & 1;
    int o    = idx >> 10;
    int k  = (e & 7) + ((lane >> 4) << 3) + ((e >> 3) << 4);
    int nn = (lane & 15) + (ct << 4);
    wpack[idx] = (_Float16)w[(o * CIN + k) * COUT + nn];
}

// Precompute sanitized neighbor index for every (offset, point):
// miss -> n (the zero feature row), hit -> min original index in that voxel.
__global__ void lookup_all_kernel(const int* __restrict__ posns,
                                  const int* __restrict__ hkey,
                                  const int* __restrict__ hval,
                                  int* __restrict__ nbTab,
                                  int n, unsigned hmask, int hshift) {
    int idx = blockIdx.x * blockDim.x + threadIdx.x;
    if (idx >= NOFF * n) return;
    int o = idx / n;
    int p = idx - o * n;
    int dx = o / 9 - 1, dy = (o / 3) % 3 - 1, dz = o % 3 - 1;
    int x = posns[p * 3 + 0], y = posns[p * 3 + 1], z = posns[p * 3 + 2];
    int kq = ((x + dx + 1) * KEYB + (y + dy + 1)) * KEYB + (z + dz + 1);
    int nb = hash_lookup(hkey, hval, kq, hmask, hshift);
    nbTab[idx] = (nb < 0) ? n : nb;
}

__device__ __forceinline__ v16h load_a_frag(const _Float16* __restrict__ featH,
                                            int nb, int c0) {
    const _Float16* f = featH + (size_t)nb * CIN + c0;
    v8h lo = *(const v8h*)f;          // K = c0    .. c0+7
    v8h hi = *(const v8h*)(f + 16);   // K = c0+16 .. c0+23
    return __builtin_shufflevector(lo, hi, 0,1,2,3,4,5,6,7,
                                           8,9,10,11,12,13,14,15);
}

// ---------------------------------------------------------------------------
// Main kernel: 1 wave = 32 points (two 16-row A tiles share the B fragments).
// Branchless per-offset body: nbTab load -> 2x b128 gather -> 4 WMMAs.
// ---------------------------------------------------------------------------
__global__ void spconv_wmma_kernel(const _Float16* __restrict__ featH,
                                   const int* __restrict__ posns,
                                   const _Float16* __restrict__ wpack,
                                   const int* __restrict__ hkey,
                                   const int* __restrict__ hval,
                                   const int* __restrict__ nbTab,
                                   float* __restrict__ out,
                                   int n, unsigned hmask, int hshift) {
    __shared__ __align__(32) _Float16 sW[WPACK_ELEMS];
    {   // cooperative 128-bit copy of packed B fragments into LDS
        const uint4* src = (const uint4*)wpack;
        uint4* dst = (uint4*)sW;
        const int cnt = WPACK_ELEMS * 2 / 16;   // 3456
        for (int i = threadIdx.x; i < cnt; i += blockDim.x) dst[i] = src[i];
    }
    __syncthreads();

    const int lane = threadIdx.x & 31;
    const int wave = threadIdx.x >> 5;
    const int tile = blockIdx.x * (blockDim.x >> 5) + wave;
    const int ntiles = (n + 31) >> 5;
    if (tile >= ntiles) return;            // wave-uniform: EXEC stays all-1s

    const int row  = lane & 15;            // matrix row handled by this lane
    const int half = lane >> 4;            // which K-half of the A fragment
    const int base = tile * 32;
    const int p0   = base + row;           // rows of A-tile 0
    const int p1   = base + 16 + row;      // rows of A-tile 1
    const int pc0  = (p0 < n) ? p0 : n - 1;
    const int pc1  = (p1 < n) ? p1 : n - 1;
    const int c0   = half * 8;             // A frag: elems 0..7  = K c0..c0+7
                                           //         elems 8..15 = K c0+16..c0+23
    int px0 = 0, py0 = 0, pz0 = 0, px1 = 0, py1 = 0, pz1 = 0;
    if (!nbTab) {
        px0 = posns[pc0*3+0]; py0 = posns[pc0*3+1]; pz0 = posns[pc0*3+2];
        px1 = posns[pc1*3+0]; py1 = posns[pc1*3+1]; pz1 = posns[pc1*3+2];
    }

    v8f acc00 = {}, acc01 = {};            // A-tile 0 x (Bct0, Bct1)
    v8f acc10 = {}, acc11 = {};            // A-tile 1 x (Bct0, Bct1)
    const v16h* sWv = (const v16h*)sW;

    #pragma unroll 3
    for (int o = 0; o < NOFF; ++o) {
        int nb0, nb1;
        if (nbTab) {
            nb0 = nbTab[o * n + pc0];
            nb1 = nbTab[o * n + pc1];
        } else {
            int dx = o / 9 - 1, dy = (o / 3) % 3 - 1, dz = o % 3 - 1;
            int k0 = ((px0 + dx + 1) * KEYB + (py0 + dy + 1)) * KEYB + (pz0 + dz + 1);
            int k1 = ((px1 + dx + 1) * KEYB + (py1 + dy + 1)) * KEYB + (pz1 + dz + 1);
            nb0 = hash_lookup(hkey, hval, k0, hmask, hshift);
            nb1 = hash_lookup(hkey, hval, k1, hmask, hshift);
            nb0 = (nb0 < 0) ? n : nb0;
            nb1 = (nb1 < 0) ? n : nb1;
        }
        v16h a0 = load_a_frag(featH, nb0, c0);
        v16h a1 = load_a_frag(featH, nb1, c0);
        v16h b0 = sWv[(o * 2 + 0) * 32 + lane];
        v16h b1 = sWv[(o * 2 + 1) * 32 + lane];
        acc00 = __builtin_amdgcn_wmma_f32_16x16x32_f16(false, a0, false, b0,
                                                       (short)0, acc00, false, false);
        acc01 = __builtin_amdgcn_wmma_f32_16x16x32_f16(false, a0, false, b1,
                                                       (short)0, acc01, false, false);
        acc10 = __builtin_amdgcn_wmma_f32_16x16x32_f16(false, a1, false, b0,
                                                       (short)0, acc10, false, false);
        acc11 = __builtin_amdgcn_wmma_f32_16x16x32_f16(false, a1, false, b1,
                                                       (short)0, acc11, false, false);
    }

    // D layout: lane(0-15)=col, VGPR m -> row m (+8 for lanes 16-31)
    const int r0 = base + half * 8;        // A-tile 0 rows
    const int r1 = base + 16 + half * 8;   // A-tile 1 rows
    #pragma unroll
    for (int m = 0; m < 8; ++m) {
        int ra = r0 + m;
        if (ra < n) {
            out[(size_t)ra * COUT + row]      = acc00[m];
            out[(size_t)ra * COUT + row + 16] = acc01[m];
        }
        int rb = r1 + m;
        if (rb < n) {
            out[(size_t)rb * COUT + row]      = acc10[m];
            out[(size_t)rb * COUT + row + 16] = acc11[m];
        }
    }
}

// ---------------------------------------------------------------------------
extern "C" void kernel_launch(void* const* d_in, const int* in_sizes, int n_in,
                              void* d_out, int out_size, void* d_ws, size_t ws_size,
                              hipStream_t stream) {
    const float* features  = (const float*)d_in[0];
    const int*   positions = (const int*)d_in[1];
    const float* weight    = (const float*)d_in[2];
    float* out = (float*)d_out;
    const int n = in_sizes[1] / 3;

    // workspace layout (256B aligned chunks)
    int  logcap = 18;
    bool withNb = true;
    size_t o_hkey = 0, o_hval = 0, o_feat = 0, o_wp = 0, o_nb = 0, total = 0;
    auto layout = [&](int lc, bool nbtab) {
        size_t off = 0;
        auto take = [&](size_t b) {
            size_t pp = off; off = (off + b + 255) & ~(size_t)255; return pp;
        };
        size_t hcap = (size_t)1 << lc;
        o_hkey = take(hcap * 4);
        o_hval = take(hcap * 4);
        o_feat = take((size_t)(n + 1) * CIN * 2);   // +1 zero row
        o_wp   = take((size_t)WPACK_ELEMS * 2);
        o_nb   = nbtab ? take((size_t)NOFF * n * 4) : 0;
        total  = off;
    };
    layout(logcap, withNb);
    if (total > ws_size) { withNb = false; layout(logcap, withNb); }
    if (total > ws_size) { logcap = 17;    layout(logcap, withNb); }

    char* ws = (char*)d_ws;
    int*      hkey  = (int*)(ws + o_hkey);
    int*      hval  = (int*)(ws + o_hval);
    _Float16* featH = (_Float16*)(ws + o_feat);
    _Float16* wpack = (_Float16*)(ws + o_wp);
    int*      nbTab = withNb ? (int*)(ws + o_nb) : nullptr;

    const int      hcap   = 1 << logcap;
    const unsigned hmask  = (unsigned)(hcap - 1);
    const int      hshift = 32 - logcap;

    init_hash_kernel<<<(hcap + 255) / 256, 256, 0, stream>>>(hkey, hval, hcap);
    build_hash_kernel<<<(n + 255) / 256, 256, 0, stream>>>(positions, hkey, hval,
                                                           n, hmask, hshift);
    const int ncvt = (n + 1) * CIN;
    cvt_feat_kernel<<<(ncvt + 255) / 256, 256, 0, stream>>>(features, featH,
                                                            ncvt, n * CIN);
    pack_w_kernel<<<(WPACK_ELEMS + 255) / 256, 256, 0, stream>>>(weight, wpack);
    if (withNb) {
        lookup_all_kernel<<<(NOFF * n + 255) / 256, 256, 0, stream>>>(
            positions, hkey, hval, nbTab, n, hmask, hshift);
    }

    const int ntiles = (n + 31) / 32;
    const int blocks = (ntiles + 7) / 8;   // 8 waves (32-pt tiles) per block
    spconv_wmma_kernel<<<blocks, 256, 0, stream>>>(featH, positions, wpack,
                                                   hkey, hval, nbTab, out,
                                                   n, hmask, hshift);
}
